// MoE_52965536694320
// MI455X (gfx1250) — matrix operand order repaired and checked by reading, not compiled, not run on hardware
//
#include <hip/hip_runtime.h>
#include <hip/hip_bf16.h>

#define N_EMBD      1024
#define MOE_HIDDEN  2048
#define N_ROUTED    7
#define NTOK        8192   // B*T = 4*2048

typedef __attribute__((ext_vector_type(16))) __bf16 v16bf;
typedef __attribute__((ext_vector_type(8)))  float  v8f;

union FragAB {
    v16bf v;
    uint4 q[2];
    unsigned short h[16];
};

__device__ __forceinline__ unsigned short f32_to_bf16(float f) {
    union { float f; unsigned int u; } c; c.f = f;
    unsigned int u = c.u + 0x7FFFu + ((c.u >> 16) & 1u);   // round-to-nearest-even
    return (unsigned short)(u >> 16);
}

__device__ __forceinline__ unsigned int pack_bf16x2(float a, float b) {
    return (unsigned int)f32_to_bf16(a) | ((unsigned int)f32_to_bf16(b) << 16);
}

__device__ __forceinline__ v8f v8f_zero() {
    v8f z = {0.f, 0.f, 0.f, 0.f, 0.f, 0.f, 0.f, 0.f};
    return z;
}

// ---------------------------------------------------------------------------
// 1. Gating: one wave per token. logits = x @ gate_w^T, softmax, argmax(p+bias)
// ---------------------------------------------------------------------------
__global__ __launch_bounds__(256)
void gate_kernel(const float* __restrict__ x, const float* __restrict__ gate_w,
                 const float* __restrict__ bias, int* __restrict__ top1,
                 float* __restrict__ gateval) {
    const int tok  = blockIdx.x * 8 + (threadIdx.x >> 5);
    const int lane = threadIdx.x & 31;
    float acc[N_ROUTED];
#pragma unroll
    for (int e = 0; e < N_ROUTED; ++e) acc[e] = 0.f;

    const float* xr = x + (size_t)tok * N_EMBD;
    for (int k = lane; k < N_EMBD; k += 32) {
        const float xv = xr[k];
#pragma unroll
        for (int e = 0; e < N_ROUTED; ++e)
            acc[e] += xv * gate_w[e * N_EMBD + k];
    }
#pragma unroll
    for (int e = 0; e < N_ROUTED; ++e)
#pragma unroll
        for (int off = 16; off > 0; off >>= 1)
            acc[e] += __shfl_xor(acc[e], off, 32);

    if (lane == 0) {
        float mx = acc[0];
#pragma unroll
        for (int e = 1; e < N_ROUTED; ++e) mx = fmaxf(mx, acc[e]);
        float p[N_ROUTED], se = 0.f;
#pragma unroll
        for (int e = 0; e < N_ROUTED; ++e) { p[e] = __expf(acc[e] - mx); se += p[e]; }
        const float inv = 1.f / se;
        int best = 0;
        float bb = p[0] * inv + bias[0];
#pragma unroll
        for (int e = 1; e < N_ROUTED; ++e) {
            const float v = p[e] * inv + bias[e];
            if (v > bb) { bb = v; best = e; }
        }
        top1[tok]    = best;
        gateval[tok] = p[best] * inv;
    }
}

// ---------------------------------------------------------------------------
// 2. Routing bookkeeping: counts -> exclusive scan -> scatter permutation
// ---------------------------------------------------------------------------
__global__ void route_init(int* counts, int* cursor) {
    if (threadIdx.x < N_ROUTED) { counts[threadIdx.x] = 0; cursor[threadIdx.x] = 0; }
}

__global__ __launch_bounds__(256)
void route_count(const int* __restrict__ top1, int* __restrict__ counts) {
    const int t = blockIdx.x * 256 + threadIdx.x;
    if (t < NTOK) atomicAdd(&counts[top1[t]], 1);
}

__global__ void route_scan(const int* __restrict__ counts, int* __restrict__ offsets,
                           int* __restrict__ cursor, float* __restrict__ aux) {
    if (threadIdx.x == 0) {
        int s = 0;
        for (int e = 0; e < N_ROUTED; ++e) {
            offsets[e] = s; cursor[e] = s; s += counts[e];
        }
        offsets[N_ROUTED] = s;
        aux[0] = 0.f;                      // aux-free strategy: loss is exactly 0
    }
}

__global__ __launch_bounds__(256)
void route_scatter(const int* __restrict__ top1, int* __restrict__ cursor,
                   int* __restrict__ perm) {
    const int t = blockIdx.x * 256 + threadIdx.x;
    if (t < NTOK) {
        const int e = top1[t];
        const int p = atomicAdd(&cursor[e], 1);
        perm[p] = t;
    }
}

// ---------------------------------------------------------------------------
// 3. Gather permuted tokens as bf16 rows (Xperm[r] = bf16(x[perm[r]]))
// ---------------------------------------------------------------------------
__global__ __launch_bounds__(256)
void gather_x(const float* __restrict__ x, const int* __restrict__ perm,
              unsigned short* __restrict__ xperm) {
    const int idx = blockIdx.x * 256 + threadIdx.x;   // one row per block, 4 elems/thread
    const int r   = idx >> 8;
    const int c   = (idx & 255) * 4;
    const int t   = perm[r];
    const float4 v = *(const float4*)(x + (size_t)t * N_EMBD + c);
    unsigned int o[2] = { pack_bf16x2(v.x, v.y), pack_bf16x2(v.z, v.w) };
    *(uint2*)(xperm + (size_t)r * N_EMBD + c) = *(const uint2*)o;
}

// ---------------------------------------------------------------------------
// 4. Fused up+gate GEMM with SiLU epilogue.
//    Block tile 128(M) x 64(N); 8 waves as 4Mx2N, each wave 32x32 per matrix.
//    Register double-buffer: prefetch tile s+1 from global while WMMAs chew s.
//    H[rows x MOE_HIDDEN] (bf16) = silu(A @ Wg) * (A @ Wu)
// ---------------------------------------------------------------------------
template <bool AF32, bool ROUTED>
__global__ __launch_bounds__(256)
void moe_up(const float* __restrict__ xf32, const unsigned short* __restrict__ xbf,
            const float* __restrict__ wg_all, const float* __restrict__ wu_all,
            const int* __restrict__ offsets, unsigned short* __restrict__ hout) {
    __shared__ unsigned short As[128 * 40];
    __shared__ unsigned short Bg[64 * 40];
    __shared__ unsigned short Bu[64 * 40];

    const int n0 = blockIdx.x * 64;
    int rowStart, rowEnd;
    const float *wg, *wu;
    if (ROUTED) {
        const int e = blockIdx.z;
        rowStart = offsets[e] + blockIdx.y * 128;
        rowEnd   = offsets[e + 1];
        if (rowStart >= rowEnd) return;
        wg = wg_all + (size_t)e * N_EMBD * MOE_HIDDEN;
        wu = wu_all + (size_t)e * N_EMBD * MOE_HIDDEN;
    } else {
        rowStart = blockIdx.y * 128;
        rowEnd   = NTOK;
        wg = wg_all; wu = wu_all;
    }

    const int tid    = threadIdx.x;
    const int lane   = tid & 31;
    const int wave   = tid >> 5;
    const int laneHi = lane >> 4;
    const int lane15 = lane & 15;
    const int waveM  = wave >> 1;     // 0..3
    const int waveN  = wave & 1;      // 0..1

    v8f G[2][2], U[2][2];
#pragma unroll
    for (int mi = 0; mi < 2; ++mi)
#pragma unroll
        for (int ni = 0; ni < 2; ++ni) { G[mi][ni] = v8f_zero(); U[mi][ni] = v8f_zero(); }

    // A staging: 2 threads/row, 16 cols each. B staging: n = tid&63, k = (tid>>6)+4j
    const int aRow = tid >> 1;
    const int aCol = (tid & 1) * 16;
    int aSrc = rowStart + aRow; if (aSrc >= rowEnd) aSrc = rowEnd - 1;
    const int bN  = tid & 63;
    const int bK0 = tid >> 6;

    float4 aS[4];          // f32 A staging regs
    uint4  aS2[2];         // bf16 A staging regs
    float  bgS[8], buS[8]; // B staging regs

    auto loadTile = [&](int s) {
        const int k0 = s * 32;
        if (AF32) {
            const float4* src = (const float4*)(xf32 + (size_t)aSrc * N_EMBD + k0 + aCol);
#pragma unroll
            for (int j = 0; j < 4; ++j) aS[j] = src[j];
        } else {
            const uint4* src = (const uint4*)(xbf + (size_t)aSrc * N_EMBD + k0 + aCol);
            aS2[0] = src[0]; aS2[1] = src[1];
        }
#pragma unroll
        for (int j = 0; j < 8; ++j) {
            const int k = bK0 + j * 4;
            bgS[j] = wg[(size_t)(k0 + k) * MOE_HIDDEN + n0 + bN];
            buS[j] = wu[(size_t)(k0 + k) * MOE_HIDDEN + n0 + bN];
        }
    };
    auto storeTile = [&]() {
        if (AF32) {
            unsigned int* dst = (unsigned int*)&As[aRow * 40 + aCol];
#pragma unroll
            for (int j = 0; j < 4; ++j) {
                dst[2 * j + 0] = pack_bf16x2(aS[j].x, aS[j].y);
                dst[2 * j + 1] = pack_bf16x2(aS[j].z, aS[j].w);
            }
        } else {
            uint4* dst = (uint4*)&As[aRow * 40 + aCol];
            dst[0] = aS2[0]; dst[1] = aS2[1];
        }
#pragma unroll
        for (int j = 0; j < 8; ++j) {
            const int k = bK0 + j * 4;
            Bg[bN * 40 + k] = f32_to_bf16(bgS[j]);
            Bu[bN * 40 + k] = f32_to_bf16(buS[j]);
        }
    };

    constexpr int S = N_EMBD / 32;
    loadTile(0);
    for (int s = 0; s < S; ++s) {
        storeTile();
        __syncthreads();
        if (s + 1 < S) loadTile(s + 1);   // overlap global latency with WMMAs below

        FragAB af[2];
#pragma unroll
        for (int mi = 0; mi < 2; ++mi) {
            const unsigned short* p =
                &As[(waveM * 32 + mi * 16 + lane15) * 40 + laneHi * 16];
            af[mi].q[0] = *(const uint4*)p;
            af[mi].q[1] = *(const uint4*)(p + 8);
        }
#pragma unroll
        for (int ni = 0; ni < 2; ++ni) {
            FragAB fg, fu;
            const unsigned short* pg =
                &Bg[(waveN * 32 + ni * 16 + lane15) * 40 + laneHi * 16];
            fg.q[0] = *(const uint4*)pg; fg.q[1] = *(const uint4*)(pg + 8);
            const unsigned short* pu =
                &Bu[(waveN * 32 + ni * 16 + lane15) * 40 + laneHi * 16];
            fu.q[0] = *(const uint4*)pu; fu.q[1] = *(const uint4*)(pu + 8);
#pragma unroll
            for (int mi = 0; mi < 2; ++mi) {
                G[mi][ni] = __builtin_amdgcn_wmma_f32_16x16x32_bf16(
                    false, af[mi].v, false, fg.v, (short)0, G[mi][ni], false, false);
                U[mi][ni] = __builtin_amdgcn_wmma_f32_16x16x32_bf16(
                    false, af[mi].v, false, fu.v, (short)0, U[mi][ni], false, false);
            }
        }
        __syncthreads();
    }

    // ---- SiLU(G) * U epilogue (v_rcp instead of IEEE divide), bf16 store ----
#pragma unroll
    for (int mi = 0; mi < 2; ++mi)
#pragma unroll
        for (int ni = 0; ni < 2; ++ni) {
            const int col = n0 + waveN * 32 + ni * 16 + lane15;
#pragma unroll
            for (int i = 0; i < 8; ++i) {
                const int row = rowStart + waveM * 32 + mi * 16 + laneHi * 8 + i;
                if (row < rowEnd) {
                    const float g   = G[mi][ni][i];
                    const float u   = U[mi][ni][i];
                    const float sig = __builtin_amdgcn_rcpf(1.f + __expf(-g));
                    hout[(size_t)row * MOE_HIDDEN + col] = f32_to_bf16(g * sig * u);
                }
            }
        }
}

// ---------------------------------------------------------------------------
// 5. Down GEMM: Y = H @ Wd.  Shared: plain store. Routed: y[t] += gv[t]*acc.
// ---------------------------------------------------------------------------
template <bool ROUTED>
__global__ __launch_bounds__(256)
void moe_down(const unsigned short* __restrict__ hbuf, const float* __restrict__ wd_all,
              const int* __restrict__ offsets, const int* __restrict__ perm,
              const float* __restrict__ gateval, float* __restrict__ out) {
    __shared__ unsigned short As[128 * 40];
    __shared__ unsigned short Bs[64 * 40];

    const int n0 = blockIdx.x * 64;
    int rowStart, rowEnd;
    const float* wd;
    if (ROUTED) {
        const int e = blockIdx.z;
        rowStart = offsets[e] + blockIdx.y * 128;
        rowEnd   = offsets[e + 1];
        if (rowStart >= rowEnd) return;
        wd = wd_all + (size_t)e * MOE_HIDDEN * N_EMBD;
    } else {
        rowStart = blockIdx.y * 128;
        rowEnd   = NTOK;
        wd = wd_all;
    }

    const int tid    = threadIdx.x;
    const int lane   = tid & 31;
    const int wave   = tid >> 5;
    const int laneHi = lane >> 4;
    const int lane15 = lane & 15;
    const int waveM  = wave >> 1;
    const int waveN  = wave & 1;

    v8f acc[2][2];
#pragma unroll
    for (int mi = 0; mi < 2; ++mi)
#pragma unroll
        for (int ni = 0; ni < 2; ++ni) acc[mi][ni] = v8f_zero();

    const int aRow = tid >> 1;
    const int aCol = (tid & 1) * 16;
    int aSrc = rowStart + aRow; if (aSrc >= rowEnd) aSrc = rowEnd - 1;
    const int bN  = tid & 63;
    const int bK0 = tid >> 6;

    uint4 aS2[2];
    float bS[8];

    auto loadTile = [&](int s) {
        const int k0 = s * 32;
        const uint4* src = (const uint4*)(hbuf + (size_t)aSrc * MOE_HIDDEN + k0 + aCol);
        aS2[0] = src[0]; aS2[1] = src[1];
#pragma unroll
        for (int j = 0; j < 8; ++j) {
            const int k = bK0 + j * 4;
            bS[j] = wd[(size_t)(k0 + k) * N_EMBD + n0 + bN];
        }
    };
    auto storeTile = [&]() {
        uint4* dst = (uint4*)&As[aRow * 40 + aCol];
        dst[0] = aS2[0]; dst[1] = aS2[1];
#pragma unroll
        for (int j = 0; j < 8; ++j) {
            const int k = bK0 + j * 4;
            Bs[bN * 40 + k] = f32_to_bf16(bS[j]);
        }
    };

    constexpr int S = MOE_HIDDEN / 32;
    loadTile(0);
    for (int s = 0; s < S; ++s) {
        storeTile();
        __syncthreads();
        if (s + 1 < S) loadTile(s + 1);

        FragAB af[2];
#pragma unroll
        for (int mi = 0; mi < 2; ++mi) {
            const unsigned short* p =
                &As[(waveM * 32 + mi * 16 + lane15) * 40 + laneHi * 16];
            af[mi].q[0] = *(const uint4*)p;
            af[mi].q[1] = *(const uint4*)(p + 8);
        }
#pragma unroll
        for (int ni = 0; ni < 2; ++ni) {
            FragAB fb;
            const unsigned short* pb =
                &Bs[(waveN * 32 + ni * 16 + lane15) * 40 + laneHi * 16];
            fb.q[0] = *(const uint4*)pb; fb.q[1] = *(const uint4*)(pb + 8);
#pragma unroll
            for (int mi = 0; mi < 2; ++mi)
                acc[mi][ni] = __builtin_amdgcn_wmma_f32_16x16x32_bf16(
                    false, af[mi].v, false, fb.v, (short)0, acc[mi][ni], false, false);
        }
        __syncthreads();
    }

#pragma unroll
    for (int mi = 0; mi < 2; ++mi)
#pragma unroll
        for (int ni = 0; ni < 2; ++ni) {
            const int col = n0 + waveN * 32 + ni * 16 + lane15;
#pragma unroll
            for (int i = 0; i < 8; ++i) {
                const int row = rowStart + waveM * 32 + mi * 16 + laneHi * 8 + i;
                if (ROUTED) {
                    if (row < rowEnd) {
                        const int t    = perm[row];
                        const float gv = gateval[t];
                        float* o = out + (size_t)t * N_EMBD + col;
                        *o = *o + gv * acc[mi][ni][i];   // race-free: top-1 owner
                    }
                } else {
                    out[(size_t)row * N_EMBD + col] = acc[mi][ni][i];
                }
            }
        }
}

// ---------------------------------------------------------------------------
extern "C" void kernel_launch(void* const* d_in, const int* in_sizes, int n_in,
                              void* d_out, int out_size, void* d_ws, size_t ws_size,
                              hipStream_t stream) {
    const float* x         = (const float*)d_in[0];
    const float* gate_w    = (const float*)d_in[1];
    const float* exp_bias  = (const float*)d_in[2];
    const float* shared_wg = (const float*)d_in[3];
    const float* shared_wu = (const float*)d_in[4];
    const float* shared_wd = (const float*)d_in[5];
    const float* routed_wg = (const float*)d_in[6];
    const float* routed_wu = (const float*)d_in[7];
    const float* routed_wd = (const float*)d_in[8];
    float* out = (float*)d_out;                       // [NTOK*N_EMBD] y, then [1] aux
    float* aux = out + (size_t)NTOK * N_EMBD;

    // workspace carve-up (256B aligned slabs)
    char* ws = (char*)d_ws;
    int*   top1    = (int*)ws;                 ws += 256 * ((NTOK * 4 + 255) / 256);
    float* gateval = (float*)ws;               ws += 256 * ((NTOK * 4 + 255) / 256);
    int*   counts  = (int*)ws;                 // 7
    int*   cursor  = counts + 8;               // 7
    int*   offsets = counts + 16;              // 8
    ws += 256;
    int*   perm    = (int*)ws;                 ws += 256 * ((NTOK * 4 + 255) / 256);
    unsigned short* xperm = (unsigned short*)ws;
    ws += (size_t)NTOK * N_EMBD * 2;           // 16 MB
    unsigned short* hbuf = (unsigned short*)ws; // 32 MB, reused shared->routed

    // 1. routing decision
    gate_kernel<<<NTOK / 8, 256, 0, stream>>>(x, gate_w, exp_bias, top1, gateval);
    // 2. permutation build
    route_init<<<1, 32, 0, stream>>>(counts, cursor);
    route_count<<<NTOK / 256, 256, 0, stream>>>(top1, counts);
    route_scan<<<1, 1, 0, stream>>>(counts, offsets, cursor, aux);
    route_scatter<<<NTOK / 256, 256, 0, stream>>>(top1, cursor, perm);
    // 3. gather permuted tokens (bf16)
    gather_x<<<NTOK, 256, 0, stream>>>(x, perm, xperm);

    // 4. shared expert: up+gate then down (plain store covers all of y)
    moe_up<true, false><<<dim3(MOE_HIDDEN / 64, NTOK / 128), 256, 0, stream>>>(
        x, nullptr, shared_wg, shared_wu, nullptr, hbuf);
    moe_down<false><<<dim3(N_EMBD / 64, NTOK / 128), 256, 0, stream>>>(
        hbuf, shared_wd, nullptr, nullptr, nullptr, out);

    // 5. routed experts: per-expert dense GEMM over gathered rows, early-exit grid
    moe_up<false, true><<<dim3(MOE_HIDDEN / 64, NTOK / 128, N_ROUTED), 256, 0, stream>>>(
        nullptr, xperm, routed_wg, routed_wu, offsets, hbuf);
    moe_down<true><<<dim3(N_EMBD / 64, NTOK / 128, N_ROUTED), 256, 0, stream>>>(
        hbuf, routed_wd, offsets, perm, gateval, out);
}
